// ANIMAApex_86887188398379
// MI455X (gfx1250) — matrix operand order repaired and checked by reading, not compiled, not run on hardware
//
#include <hip/hip_runtime.h>

#define DEV __device__ __forceinline__

typedef __attribute__((ext_vector_type(16))) _Float16 v16h;
typedef __attribute__((ext_vector_type(8)))  float    v8f;

#if defined(__has_builtin)
#if __has_builtin(__builtin_amdgcn_tanhf)
#define ANIMA_HAS_TANH 1
#endif
#endif

namespace {

constexpr int Tn = 2048; // seq len
constexpr int Bn = 1024; // batch
constexpr int Sn = 8;    // sensory dim
constexpr int On = 4;    // output dim

struct Params {
  const float *x;
  const float *W_enc_w, *W_enc_b;
  const float *W_from_W, *W_from_I, *W_from_A;
  const float *W_gate_w, *W_gate_b;
  const float *I_z_w, *I_z_b, *I_r_w, *I_r_b, *I_h_w, *I_h_b;
  const float *cond_w, *cond_b;
  const float *then_w, *then_b, *else_w, *else_b;
  const float *A_from_W, *A_from_I, *A_from_A;
  const float *A_gate_w, *A_gate_b;
  const float *phi_w, *phi_b;
  float *out;
};

union U16 { v16h h; unsigned u[8]; };

DEV unsigned pkh(float a, float b) {
  return __builtin_bit_cast(unsigned, __builtin_amdgcn_cvt_pkrtz(a, b));
}

#ifdef ANIMA_HAS_TANH
DEV float tanh_(float x) { return __builtin_amdgcn_tanhf(x); }       // v_tanh_f32
DEV float sigm(float x)  { return __builtin_fmaf(0.5f, __builtin_amdgcn_tanhf(0.5f * x), 0.5f); }
#else
DEV float sigm(float x) {
  return __builtin_amdgcn_rcpf(1.0f + __builtin_amdgcn_exp2f(-1.4426950408889634f * x));
}
DEV float tanh_(float x) { return 2.0f * sigm(2.0f * x) - 1.0f; }
#endif

DEV v8f splat8(float v) {
  v8f r;
#pragma unroll
  for (int i = 0; i < 8; ++i) r[i] = v;
  return r;
}

// ---- Weight A-fragment loader (K-PERMUTED, bias-folding) -------------------
// K relabeling so B-operands need no cross-lane movement:
//   q 0..7   = slot1 dims 0..7     q 8..15  = slot2 dims 0..7
//   q 16..23 = slot1 dims 8..15    q 24..31 = slot2 dims 8..15
// A element e at lane-half hi: q = e + 16*hi,
//   source = (q&8) ? W2 : W1 ; row = (q&7) + ((q&16)>>1)
// Bias folding: pass W2 = bias vector with rlim2 = 1 -> bias lands in A column
// q=8 (row 0). The matching B operand supplies 1.0 at K=8 (u[4] low half,
// lanes 0-15), so D accumulates bias[m] for every batch column.
DEV v16h mk_frag(const float* W1, int rlim1, const float* W2, int rlim2,
                 int ncols, int m, int hi, int mlim) {
  v16h f;
#pragma unroll
  for (int e = 0; e < 16; ++e) {
    int q   = e + 16 * hi;
    int row = (q & 7) + ((q & 16) >> 1);
    float v = 0.0f;
    if (m < mlim) {
      if (q & 8) { if (W2 && row < rlim2) v = W2[row * ncols + m]; }
      else       { if (W1 && row < rlim1) v = W1[row * ncols + m]; }
    }
    f[e] = (_Float16)v;
  }
  return f;
}

// Bias in C/D layout (for the two full-K gate matmuls that can't fold).
DEV v8f mk_bias(const float* b, int n, int hi) {
  v8f r;
#pragma unroll
  for (int v = 0; v < 8; ++v) {
    int i = v + 8 * hi;
    r[v] = (i < n) ? b[i] : 0.0f;
  }
  return r;
}

// ---- State plumbing --------------------------------------------------------
// C-layout state: lane n holds dims 0-7 (f32 c[v]=dim v), lane n+16 dims 8-15.
// Packed p[j] = f16x2 of local dims {2j,2j+1} == correct B content directly.
struct PKS { unsigned p[4]; };

DEV PKS packState(const v8f c) {
  PKS s;
#pragma unroll
  for (int j = 0; j < 4; ++j) s.p[j] = pkh(c[2 * j], c[2 * j + 1]);
  return s;
}

// Composite [S1;S2] B operand: slot1 halves then slot2 halves. No shuffles.
DEV v16h makeB2(const PKS& a, const PKS& b) {
  U16 r;
#pragma unroll
  for (int j = 0; j < 4; ++j) { r.u[j] = a.p[j]; r.u[j + 4] = b.p[j]; }
  return r.h;
}
// Single state in slot1; slot2 carries the bias channel: 1.0 at K=8.
DEV v16h makeB1(const PKS& a) {
  U16 r;
#pragma unroll
  for (int j = 0; j < 4; ++j) r.u[j] = a.p[j];
  r.u[4] = 0x00003C00u;  // f16 {1.0, 0}
  r.u[5] = 0u; r.u[6] = 0u; r.u[7] = 0u;
  return r.h;
}

DEV v8f wmma(v16h a, v16h b, v8f c) {
  return __builtin_amdgcn_wmma_f32_16x16x32_f16(
      /*neg_a=*/false, a, /*neg_b=*/false, b,
      /*c_mod=*/(short)0, c, /*reuse_a=*/false, /*reuse_b=*/false);
}

__global__ __launch_bounds__(32)
void anima_scan_kernel(Params p) {
  const int lane = threadIdx.x & 31;
  const int tile = blockIdx.x;           // 16 batch rows per wave
  const int m    = lane & 15;
  const int hi   = lane >> 4;

  // ---- Weights as A fragments (transposed GEMM: M=out-dim, K=permuted in) --
  v16h Aenc  = mk_frag(p.W_enc_w, 8,  p.W_enc_b, 1, 16, m, hi, 16);          // [x | bias]
  v16h AWfWI = mk_frag(p.W_from_W, 16, p.W_from_I, 16, 16, m, hi, 16);       // [W;I]
  v16h AWfA  = mk_frag(p.W_from_A, 16, nullptr, 0, 16, m, hi, 16);           // [A]
  v16h AWg   = mk_frag(p.W_gate_w, 16, p.W_gate_w + 256, 16, 16, m, hi, 16); // [I;A]
  v16h Az1   = mk_frag(p.I_z_w, 16, p.I_z_w + 256, 16, 16, m, hi, 16);       // [Wn;I]
  v16h Az2   = mk_frag(p.I_z_w + 512, 16, p.I_z_b, 1, 16, m, hi, 16);        // [A | bias]
  v16h Ar1   = mk_frag(p.I_r_w, 16, p.I_r_w + 256, 16, 16, m, hi, 16);
  v16h Ar2   = mk_frag(p.I_r_w + 512, 16, p.I_r_b, 1, 16, m, hi, 16);        // [A | bias]
  v16h Ah1   = mk_frag(p.I_h_w, 16, p.I_h_w + 256, 16, 16, m, hi, 16);       // [Wn;rI]
  v16h Ah2   = mk_frag(p.I_h_w + 512, 16, p.I_h_b, 1, 16, m, hi, 16);        // [A | bias]
  v16h Ath   = mk_frag(p.then_w, 16, p.then_b, 1, 16, m, hi, 16);            // [In | bias]
  v16h Ael   = mk_frag(p.else_w, 16, p.else_b, 1, 16, m, hi, 16);            // [In | bias]
  v16h AAfWI = mk_frag(p.A_from_W, 16, p.A_from_I, 16, 16, m, hi, 16);       // [Wn;In]
  v16h AAfA  = mk_frag(p.A_from_A, 16, nullptr, 0, 16, m, hi, 16);           // [A]
  v16h AAg   = mk_frag(p.A_gate_w, 16, p.A_gate_w + 256, 16, 16, m, hi, 16); // [Wn;In]
  v16h Aphi  = mk_frag(p.phi_w, 16, p.phi_b, 1, 4, m, hi, 4);                // [An | bias]

  // Only the two full-K gate matmuls keep register biases.
  v8f bwg = mk_bias(p.W_gate_b, 16, hi);
  v8f bag = mk_bias(p.A_gate_b, 16, hi);

  float cw1[8], cw2[8];
#pragma unroll
  for (int v = 0; v < 8; ++v) {
    cw1[v] = p.cond_w[v + 8 * hi];        // cond_w rows 0..15  <-> W_new dims
    cw2[v] = p.cond_w[16 + v + 8 * hi];   // cond_w rows 16..31 <-> I_new dims
  }
  const float cb = p.cond_b[0];
  const v8f Z8 = splat8(0.0f);            // inline-0 C operand

  // ---- State init ----------------------------------------------------------
  v8f If = splat8(0.0f);                  // I in f32 C-layout (GRU update)
  PKS sW, sI, sA;
#pragma unroll
  for (int j = 0; j < 4; ++j) { sW.p[j] = 0u; sI.p[j] = 0u; sA.p[j] = 0u; }

  const float* xrow = p.x + ((size_t)(tile * 16 + m) * Tn) * Sn;
  float*       orow = p.out + ((size_t)(tile * 16 + m) * Tn) * On;

  float4 nx0 = *(const float4*)(xrow);
  float4 nx1 = *(const float4*)(xrow + 4);

  for (int t = 0; t < Tn; ++t) {
    float4 x0 = nx0, x1 = nx1;
    if (t + 1 < Tn) {                     // double-buffer next x_t
      nx0 = *(const float4*)(xrow + (size_t)(t + 1) * Sn);
      nx1 = *(const float4*)(xrow + (size_t)(t + 1) * Sn + 4);
    }

    // x_t as B operand: slot1 = sensory dims 0..7, bias channel 1.0 at K=8.
    U16 bx;
    bx.u[0] = pkh(x0.x, x0.y); bx.u[1] = pkh(x0.z, x0.w);
    bx.u[2] = pkh(x1.x, x1.y); bx.u[3] = pkh(x1.z, x1.w);
    bx.u[4] = 0x00003C00u; bx.u[5] = 0u; bx.u[6] = 0u; bx.u[7] = 0u;

    // ---- Front: everything that depends only on old state + x_t -----------
    v16h B_WI = makeB2(sW, sI);           // [W;I]
    v16h B_IA = makeB2(sI, sA);           // [I;A]
    v16h B_A1 = makeB1(sA);               // [A | 1]

    v8f cxe = wmma(Aenc, bx.h, Z8);       // enc + benc (folded)
    v8f cw  = wmma(AWfA, B_A1, Z8);       // A@W_from_A
    cw      = wmma(AWfWI, B_WI, cw);      // + W@Wfw + I@Wfi
    v8f cgw = wmma(AWg, B_IA, bwg);       // W-gate pre-act
    v8f cz  = wmma(Az2, B_A1, Z8);        // z: A part + bz (folded)
    v8f cr  = wmma(Ar2, B_A1, Z8);        // r: A part + br
    v8f ch  = wmma(Ah2, B_A1, Z8);        // h: A part + bh
    v8f cai = wmma(AAfA, B_A1, Z8);       // A_input: A@A_from_A

    // xe = tanh(enc); W_new = tanh(cw + xe) * sigm(gate)
    v8f Wn;
#pragma unroll
    for (int v = 0; v < 8; ++v) {
      float xe = tanh_(cxe[v]);
      Wn[v] = tanh_(cw[v] + xe) * sigm(cgw[v]);
    }
    PKS sWn = packState(Wn);

    // GRU-ish: combined = [W_new, I, A]
    v16h B_cmb = makeB2(sWn, sI);
    cz = wmma(Az1, B_cmb, cz);
    cr = wmma(Ar1, B_cmb, cr);
    v8f rI;
#pragma unroll
    for (int v = 0; v < 8; ++v) rI[v] = sigm(cr[v]) * If[v];
    PKS srI = packState(rI);
    v16h B_h = makeB2(sWn, srI);          // [W_new, r*I, A]
    ch = wmma(Ah1, B_h, ch);
    v8f In;
#pragma unroll
    for (int v = 0; v < 8; ++v) {
      float z = sigm(cz[v]);
      In[v] = If[v] + z * (tanh_(ch[v]) - If[v]);   // (1-z)I + z h
    }
    PKS sIn = packState(In);

    // cond gate: sigmoid([W_new, I_new] @ cond_w + b), shared by lane pair
    float part = 0.0f;
#pragma unroll
    for (int v = 0; v < 8; ++v) part += cw1[v] * Wn[v] + cw2[v] * In[v];
    float cgate = sigm(part + __shfl_xor(part, 16, 32) + cb);

    // branched = else + cg*(then - else), biases folded into Ath/Ael
    v16h B_In1 = makeB1(sIn);
    v8f cth = wmma(Ath, B_In1, Z8);
    v8f cel = wmma(Ael, B_In1, Z8);
    v8f cbr;
#pragma unroll
    for (int v = 0; v < 8; ++v) {
      float te = tanh_(cel[v]);
      cbr[v] = __builtin_fmaf(cgate, tanh_(cth[v]) - te, te);
    }

    // A_new = tanh(Wn@AfW + In@AfI + A@AfA + branched) * sigm(cond_in@Ag + b)
    v16h B_WnIn = makeB2(sWn, sIn);
    cai = wmma(AAfWI, B_WnIn, cai);
    v8f cag = wmma(AAg, B_WnIn, bag);
    v8f An;
#pragma unroll
    for (int v = 0; v < 8; ++v) An[v] = tanh_(cai[v] + cbr[v]) * sigm(cag[v]);
    PKS sAn = packState(An);

    // action = A_new @ phi + b (folded) : rows 0..3 of D, lanes 0..15
    v16h B_An1 = makeB1(sAn);
    v8f cph = wmma(Aphi, B_An1, Z8);
    if (hi == 0) {
      float4 o; o.x = cph[0]; o.y = cph[1]; o.z = cph[2]; o.w = cph[3];
      *(float4*)(orow + (size_t)t * On) = o;
    }

    sW = sWn; sI = sIn; sA = sAn; If = In;
  }
}

} // namespace

extern "C" void kernel_launch(void* const* d_in, const int* in_sizes, int n_in,
                              void* d_out, int out_size, void* d_ws, size_t ws_size,
                              hipStream_t stream) {
  (void)in_sizes; (void)n_in; (void)d_ws; (void)ws_size; (void)out_size;
  Params p;
  p.x        = (const float*)d_in[0];
  p.W_enc_w  = (const float*)d_in[1];  p.W_enc_b  = (const float*)d_in[2];
  p.W_from_W = (const float*)d_in[3];  p.W_from_I = (const float*)d_in[4];
  p.W_from_A = (const float*)d_in[5];
  p.W_gate_w = (const float*)d_in[6];  p.W_gate_b = (const float*)d_in[7];
  p.I_z_w    = (const float*)d_in[8];  p.I_z_b    = (const float*)d_in[9];
  p.I_r_w    = (const float*)d_in[10]; p.I_r_b    = (const float*)d_in[11];
  p.I_h_w    = (const float*)d_in[12]; p.I_h_b    = (const float*)d_in[13];
  p.cond_w   = (const float*)d_in[14]; p.cond_b   = (const float*)d_in[15];
  p.then_w   = (const float*)d_in[16]; p.then_b   = (const float*)d_in[17];
  p.else_w   = (const float*)d_in[18]; p.else_b   = (const float*)d_in[19];
  p.A_from_W = (const float*)d_in[20]; p.A_from_I = (const float*)d_in[21];
  p.A_from_A = (const float*)d_in[22];
  p.A_gate_w = (const float*)d_in[23]; p.A_gate_b = (const float*)d_in[24];
  p.phi_w    = (const float*)d_in[25]; p.phi_b    = (const float*)d_in[26];
  p.out      = (float*)d_out;

  dim3 grid(Bn / 16);   // 64 waves, one 16-batch tile each
  dim3 block(32);       // one wave32
  anima_scan_kernel<<<grid, block, 0, stream>>>(p);
}
